// DeepMambaModel_18253611008299
// MI455X (gfx1250) — matrix operand order repaired and checked by reading, not compiled, not run on hardware
//
#include <hip/hip_runtime.h>
#include <hip/hip_bf16.h>

// ---------------------------------------------------------------------------
// DeepMamba decision-transformer forward for MI455X (gfx1250, wave32, WMMA)
// B=2, L=1024, DM=512, DM4=128, DI=1024, DS=16, DTR=32, DC=4, NL=3
// GEMMs: v_wmma_f32_16x16x32_f16 with TDM (tensor_load_to_lds) A-tile staging.
// ---------------------------------------------------------------------------

#define BB   2
#define LL   1024
#define ROWS (BB * LL)      // 2048
#define DMm  512
#define DM4m 128
#define DIm  1024
#define DSs  16
#define DTRr 32
#define DCc  4
#define NLl  3
#define SDd  17
#define ADd  7

typedef __attribute__((ext_vector_type(16))) _Float16 v16h;
typedef __attribute__((ext_vector_type(8)))  float    v8f;
typedef __attribute__((ext_vector_type(2)))  _Float16 h2;
typedef __attribute__((ext_vector_type(4)))  unsigned u32x4;
typedef __attribute__((ext_vector_type(8)))  int      i32x8;
typedef __attribute__((ext_vector_type(4)))  int      i32x4;

union HFrag { v16h v; h2 p[8]; };

// ---------------------------------------------------------------------------
// f32 -> f16 weight conversion
// ---------------------------------------------------------------------------
__global__ void dm_f32_to_f16(const float* __restrict__ src,
                              _Float16* __restrict__ dst, int n) {
    int i = blockIdx.x * blockDim.x + threadIdx.x;
    if (i < n) dst[i] = (_Float16)src[i];
}

// ---------------------------------------------------------------------------
// Token embedding: x[b,l, {re,ce,se,ae}] interleave, ae shifted by one step
// grid = ROWS blocks of 128 threads (one per DM4 dim)
// ---------------------------------------------------------------------------
__global__ void dm_embed(const float* __restrict__ st,  const float* __restrict__ ac,
                         const float* __restrict__ rtg, const float* __restrict__ ctg,
                         const int*   __restrict__ ts,
                         const float* __restrict__ Ws,  const float* __restrict__ bs,
                         const float* __restrict__ Wa,  const float* __restrict__ ba,
                         const float* __restrict__ Wr,  const float* __restrict__ br,
                         const float* __restrict__ Wc,  const float* __restrict__ bc,
                         const float* __restrict__ Et,
                         float* __restrict__ x) {
    int row = blockIdx.x;
    int d   = threadIdx.x;
    int l   = row % LL;
    int t   = ts[row];
    float te = Et[(size_t)t * DM4m + d];

    float re = Wr[d] * rtg[row] + br[d] + te;
    float ce = Wc[d] * ctg[row] + bc[d] + te;
    float se = bs[d] + te;
    #pragma unroll
    for (int k = 0; k < SDd; ++k)
        se += Ws[d * SDd + k] * st[(size_t)row * SDd + k];

    float ae = 0.f;
    if (l > 0) {
        int tp = ts[row - 1];
        ae = ba[d] + Et[(size_t)tp * DM4m + d];
        #pragma unroll
        for (int k = 0; k < ADd; ++k)
            ae += Wa[d * ADd + k] * ac[(size_t)(row - 1) * ADd + k];
    }
    float* xr = x + (size_t)row * DMm;
    xr[0 * DM4m + d] = re;
    xr[1 * DM4m + d] = ce;
    xr[2 * DM4m + d] = se;
    xr[3 * DM4m + d] = ae;
}

// ---------------------------------------------------------------------------
// LayerNorm over 512, writes f32 and f16 copies. grid=ROWS, block=256
// ---------------------------------------------------------------------------
__global__ void dm_layernorm(const float* __restrict__ x,
                             const float* __restrict__ g,
                             const float* __restrict__ b,
                             float* __restrict__ y32,
                             _Float16* __restrict__ y16) {
    __shared__ float red[256];
    int row = blockIdx.x;
    int t   = threadIdx.x;
    const float* xr = x + (size_t)row * DMm;
    float v0 = xr[t], v1 = xr[t + 256];

    red[t] = v0 + v1;
    __syncthreads();
    for (int o = 128; o > 0; o >>= 1) {
        if (t < o) red[t] += red[t + o];
        __syncthreads();
    }
    float mu = red[0] * (1.f / DMm);
    __syncthreads();

    float d0 = v0 - mu, d1 = v1 - mu;
    red[t] = d0 * d0 + d1 * d1;
    __syncthreads();
    for (int o = 128; o > 0; o >>= 1) {
        if (t < o) red[t] += red[t + o];
        __syncthreads();
    }
    float rs = rsqrtf(red[0] * (1.f / DMm) + 1e-5f);

    float o0 = d0 * rs * g[t] + b[t];
    float o1 = d1 * rs * g[t + 256] + b[t + 256];
    y32[(size_t)row * DMm + t]       = o0;
    y32[(size_t)row * DMm + t + 256] = o1;
    y16[(size_t)row * DMm + t]       = (_Float16)o0;
    y16[(size_t)row * DMm + t + 256] = (_Float16)o1;
}

// ---------------------------------------------------------------------------
// Generic WMMA GEMM:  C[M,N] = A[M,K] * W[N,K]^T   (+ epilogue)
//   A row-major f16 (lda), W row-major f16, N rows of K (ldb), C f32 (ldc)
// One wave per block; wave computes a 16x64 strip.
// The 16xK A tile is staged into LDS by the Tensor Data Mover
// (tensor_load_to_lds, D# per CDNA5 ISA) and
// re-read through DS; B streams from global with unconditional prefetch.
// MODE 0: store   MODE 1: softplus(acc + bias[col])   MODE 2: C += acc
// grid = (N/64, M/16), block = 32 (one wave)
// ---------------------------------------------------------------------------
template <int MODE>
__global__ void dm_gemm_wmma(const _Float16* __restrict__ A, int lda,
                             const _Float16* __restrict__ W, int ldb,
                             float* __restrict__ C, int ldc,
                             const float* __restrict__ bias, int K) {
    __shared__ _Float16 lA[16 * 1024];   // 32 KB: 16 rows x K (K <= 1024)

    int lane = threadIdx.x & 31;
    int m0   = blockIdx.y * 16;
    int n0   = blockIdx.x * 64;

    // ---- TDM: DMA the 16xK A tile (row-major, stride lda) into LDS --------
    {
        unsigned long long ga = (unsigned long long)(const void*)(A + (size_t)m0 * lda);
        unsigned loff = (unsigned)(unsigned long long)(const void*)lA; // LDS byte offset
        u32x4 g0;
        g0.x = 1u;                                   // count=1 (valid user D#)
        g0.y = loff;                                 // lds_addr
        g0.z = (unsigned)ga;                         // global_addr[31:0]
        g0.w = ((unsigned)(ga >> 32) & 0x01FFFFFFu)  // global_addr[56:32]
               | (2u << 30);                         // type=2 ("image")
        i32x8 g1;
        g1[0] = 0x00010000;                  // workgroup_mask=0, data_size=1 (2B)
        g1[1] = (int)((unsigned)K << 16);    // tensor_dim0[15:0] (K)
        g1[2] = (int)(16u << 16);            // tensor_dim0 hi=0 | tensor_dim1=16
        g1[3] = (int)((unsigned)K << 16);    // tensor_dim1 hi=0 | tile_dim0=K
        g1[4] = 16;                          // tile_dim1=16, tile_dim2=0
        g1[5] = lda;                         // tensor_dim0_stride[31:0]
        g1[6] = 0;                           // stride hi / dim1_stride lo
        g1[7] = 0;
        i32x4 gz = {0, 0, 0, 0};             // groups 2/3 unused (2-D tensor)
        asm volatile("tensor_load_to_lds %0, %1, %2, %3"
                     :: "s"(g0), "s"(g1), "s"(gz), "s"(gz)
                     : "memory");
        __builtin_amdgcn_s_wait_tensorcnt(0);
        asm volatile("" ::: "memory");
    }

    // A fragment addressing (16x32 f16, wave32 layout):
    //  lanes 0-15: row m, K = 0..15
    //  lanes 16-31: same rows, K = 16..31
    const _Float16* aRow = lA + (size_t)(lane & 15) * K + ((lane >> 4) * 16);
    // B fragment addressing (32x16 f16): lanes 0-15 cols n, K=0..15;
    //  lanes 16-31 same cols, K=16..31. K pairs contiguous in memory.
    int nB = lane & 15;
    int kb = (lane >> 4) * 16;

    v8f acc[4] = {};

    for (int k0 = 0; k0 < K; k0 += 32) {
        HFrag a;
        const _Float16* ap = aRow + k0;
        #pragma unroll
        for (int j = 0; j < 4; ++j) {
            a.p[j]     = *(const h2*)(ap + 2 * j);        // ds_load
            a.p[4 + j] = *(const h2*)(ap + 8 + 2 * j);
        }

        #pragma unroll
        for (int t = 0; t < 4; ++t) {
            HFrag bfr;
            const _Float16* bp = W + (size_t)(n0 + t * 16 + nB) * ldb + k0 + kb;
            #pragma unroll
            for (int j = 0; j < 8; ++j) bfr.p[j] = *(const h2*)(bp + 2 * j);
            __builtin_prefetch(bp + 32, 0, 3);   // speculative global_prefetch_b8
            acc[t] = __builtin_amdgcn_wmma_f32_16x16x32_f16(
                false, a.v, false, bfr.v, (short)0, acc[t], false, false);
        }
    }

    // C layout: lane -> col n = lane&15, rows mbase..mbase+7
    int n     = lane & 15;
    int mbase = m0 + (lane >> 4) * 8;
    #pragma unroll
    for (int t = 0; t < 4; ++t) {
        int nc = n0 + t * 16 + n;
        #pragma unroll
        for (int r = 0; r < 8; ++r) {
            float v = acc[t][r];
            float* cp = C + (size_t)(mbase + r) * ldc + nc;
            if (MODE == 1) {
                v += bias[nc];
                v = (v > 20.f) ? v : log1pf(__expf(v));
                *cp = v;
            } else if (MODE == 2) {
                *cp += v;
            } else {
                *cp = v;
            }
        }
    }
}

// ---------------------------------------------------------------------------
// Causal depthwise conv (DC=4) + SiLU over first DI channels of xz.
// thread per (row, ch); grid = ROWS*DI/256
// ---------------------------------------------------------------------------
__global__ void dm_conv_silu(const float* __restrict__ xz,
                             const float* __restrict__ cw,
                             const float* __restrict__ cb,
                             float* __restrict__ xc32,
                             _Float16* __restrict__ xc16) {
    int idx = blockIdx.x * blockDim.x + threadIdx.x;
    int ch  = idx & (DIm - 1);
    int row = idx >> 10;
    int l   = row % LL;
    float acc = cb[ch];
    #pragma unroll
    for (int k = 0; k < DCc; ++k) {
        int ls = l + k - (DCc - 1);
        if (ls >= 0)
            acc += xz[(size_t)(row + k - (DCc - 1)) * (2 * DIm) + ch] * cw[ch * DCc + k];
    }
    float s = acc / (1.f + __expf(-acc));   // SiLU
    xc32[idx] = s;
    xc16[idx] = (_Float16)s;
}

// ---------------------------------------------------------------------------
// Extract dt_low (first 32 of xdbl row) -> f16 buffer for the dtproj GEMM
// ---------------------------------------------------------------------------
__global__ void dm_dtlow(const float* __restrict__ xdbl,
                         _Float16* __restrict__ dtlow) {
    int idx = blockIdx.x * blockDim.x + threadIdx.x;   // ROWS*32
    int r   = idx & (DTRr - 1);
    int row = idx >> 5;
    dtlow[idx] = (_Float16)xdbl[(size_t)row * (DTRr + 2 * DSs) + r];
}

// ---------------------------------------------------------------------------
// Selective scan. grid = B * (DI/256) = 8 blocks, 256 threads (1 channel each).
// h[16] in registers; B_t/C_t broadcast via LDS; fused D*u and SiLU(z) gating.
// Writes gated output as f16 for the out_proj WMMA GEMM.
// ---------------------------------------------------------------------------
__global__ void dm_scan(const float* __restrict__ dt,    // (ROWS, DI)
                        const float* __restrict__ xdbl,  // (ROWS, 64)
                        const float* __restrict__ xc32,  // (ROWS, DI)
                        const float* __restrict__ xz,    // (ROWS, 2*DI)
                        const float* __restrict__ Alog,  // (DI, DS)
                        const float* __restrict__ Dv,    // (DI)
                        _Float16* __restrict__ yg16) {   // (ROWS, DI)
    __shared__ float sB[DSs], sC[DSs];
    int b  = blockIdx.x >> 2;
    int ch = (blockIdx.x & 3) * 256 + threadIdx.x;

    float A[DSs];
    #pragma unroll
    for (int s = 0; s < DSs; ++s) A[s] = -__expf(Alog[(size_t)ch * DSs + s]);
    float Dc = Dv[ch];

    float h[DSs];
    #pragma unroll
    for (int s = 0; s < DSs; ++s) h[s] = 0.f;

    for (int l = 0; l < LL; ++l) {
        int row = b * LL + l;
        if (threadIdx.x < DSs) {
            const float* xr = xdbl + (size_t)row * (DTRr + 2 * DSs);
            sB[threadIdx.x] = xr[DTRr + threadIdx.x];
            sC[threadIdx.x] = xr[DTRr + DSs + threadIdx.x];
        }
        __syncthreads();

        float dtv = dt[(size_t)row * DIm + ch];
        float u   = xc32[(size_t)row * DIm + ch];
        float du  = dtv * u;
        float y   = 0.f;
        #pragma unroll
        for (int s = 0; s < DSs; ++s) {
            h[s] = h[s] * __expf(dtv * A[s]) + du * sB[s];
            y   += h[s] * sC[s];
        }
        y += Dc * u;
        float z = xz[(size_t)row * (2 * DIm) + DIm + ch];
        y *= z / (1.f + __expf(-z));        // * SiLU(z)
        yg16[(size_t)row * DIm + ch] = (_Float16)y;
        __syncthreads();
    }
}

// ---------------------------------------------------------------------------
// Output heads: state_preds (17) from slot 3, action_preds (7) from slot 2.
// grid = ROWS, block = 32 (one wave)
// ---------------------------------------------------------------------------
__global__ void dm_heads(const float* __restrict__ xf,
                         const float* __restrict__ Wps, const float* __restrict__ bps,
                         const float* __restrict__ Wpa, const float* __restrict__ bpa,
                         float* __restrict__ out) {
    int row = blockIdx.x;
    int t   = threadIdx.x;
    const float* xr = xf + (size_t)row * DMm;
    if (t < SDd) {
        float a = bps[t];
        #pragma unroll 8
        for (int k = 0; k < DM4m; ++k) a += xr[3 * DM4m + k] * Wps[t * DM4m + k];
        out[(size_t)row * SDd + t] = a;
    } else if (t < SDd + ADd) {
        int q = t - SDd;
        float a = bpa[q];
        #pragma unroll 8
        for (int k = 0; k < DM4m; ++k) a += xr[2 * DM4m + k] * Wpa[q * DM4m + k];
        out[(size_t)ROWS * SDd + (size_t)row * ADd + q] = a;
    }
}

// ---------------------------------------------------------------------------
// Host launcher
// ---------------------------------------------------------------------------
extern "C" void kernel_launch(void* const* d_in, const int* in_sizes, int n_in,
                              void* d_out, int out_size, void* d_ws, size_t ws_size,
                              hipStream_t stream) {
    (void)in_sizes; (void)n_in; (void)out_size; (void)ws_size;

    const float* states   = (const float*)d_in[0];
    const float* actions  = (const float*)d_in[1];
    const float* rtg      = (const float*)d_in[2];
    const float* ctg      = (const float*)d_in[3];
    const int*   tsteps   = (const int*)  d_in[4];
    const float* W_s      = (const float*)d_in[5];
    const float* b_s      = (const float*)d_in[6];
    const float* W_a      = (const float*)d_in[7];
    const float* b_a      = (const float*)d_in[8];
    const float* W_r      = (const float*)d_in[9];
    const float* b_r      = (const float*)d_in[10];
    const float* W_c      = (const float*)d_in[11];
    const float* b_c      = (const float*)d_in[12];
    const float* E_t      = (const float*)d_in[13];
    const float* ln_g     = (const float*)d_in[14];
    const float* ln_b     = (const float*)d_in[15];
    const float* in_w     = (const float*)d_in[16];
    const float* conv_w   = (const float*)d_in[17];
    const float* conv_b   = (const float*)d_in[18];
    const float* xproj_w  = (const float*)d_in[19];
    const float* dtproj_w = (const float*)d_in[20];
    const float* dtproj_b = (const float*)d_in[21];
    const float* A_log    = (const float*)d_in[22];
    const float* D_ssm    = (const float*)d_in[23];
    const float* out_w    = (const float*)d_in[24];
    const float* fn_g     = (const float*)d_in[25];
    const float* fn_b     = (const float*)d_in[26];
    const float* W_ps     = (const float*)d_in[27];
    const float* b_ps     = (const float*)d_in[28];
    const float* W_pa     = (const float*)d_in[29];
    const float* b_pa     = (const float*)d_in[30];

    // workspace carve-up
    char* w = (char*)d_ws;
    auto alloc = [&](size_t bytes) -> char* {
        char* p = w;
        w += (bytes + 255) & ~(size_t)255;
        return p;
    };
    float*     x      = (float*)    alloc((size_t)ROWS * DMm * 4);
    float*     xf32   = (float*)    alloc((size_t)ROWS * DMm * 4);
    _Float16*  xn16   = (_Float16*) alloc((size_t)ROWS * DMm * 2);
    float*     xz     = (float*)    alloc((size_t)ROWS * 2 * DIm * 4);
    float*     xc32   = (float*)    alloc((size_t)ROWS * DIm * 4);
    _Float16*  xc16   = (_Float16*) alloc((size_t)ROWS * DIm * 2);
    float*     xdbl   = (float*)    alloc((size_t)ROWS * (DTRr + 2 * DSs) * 4);
    _Float16*  dtlo16 = (_Float16*) alloc((size_t)ROWS * DTRr * 2);
    float*     dtb    = (float*)    alloc((size_t)ROWS * DIm * 4);
    _Float16*  yg16   = (_Float16*) alloc((size_t)ROWS * DIm * 2);
    _Float16*  w16in  = (_Float16*) alloc((size_t)NLl * 2 * DIm * DMm * 2);
    _Float16*  w16xp  = (_Float16*) alloc((size_t)NLl * (DTRr + 2 * DSs) * DIm * 2);
    _Float16*  w16dt  = (_Float16*) alloc((size_t)NLl * DIm * DTRr * 2);
    _Float16*  w16out = (_Float16*) alloc((size_t)NLl * DMm * DIm * 2);

    auto cvt = [&](const float* src, _Float16* dst, int n) {
        dm_f32_to_f16<<<(n + 255) / 256, 256, 0, stream>>>(src, dst, n);
    };
    cvt(in_w,     w16in,  NLl * 2 * DIm * DMm);
    cvt(xproj_w,  w16xp,  NLl * (DTRr + 2 * DSs) * DIm);
    cvt(dtproj_w, w16dt,  NLl * DIm * DTRr);
    cvt(out_w,    w16out, NLl * DMm * DIm);

    dm_embed<<<ROWS, DM4m, 0, stream>>>(states, actions, rtg, ctg, tsteps,
                                        W_s, b_s, W_a, b_a, W_r, b_r, W_c, b_c,
                                        E_t, x);

    for (int i = 0; i < NLl; ++i) {
        // LayerNorm
        dm_layernorm<<<ROWS, 256, 0, stream>>>(x, ln_g + i * DMm, ln_b + i * DMm,
                                               xf32, xn16);
        // in_proj: (2048,512) x (2048,512)^T -> (2048,2048)
        dm_gemm_wmma<0><<<dim3((2 * DIm) / 64, ROWS / 16), 32, 0, stream>>>(
            xn16, DMm, w16in + (size_t)i * 2 * DIm * DMm, DMm,
            xz, 2 * DIm, nullptr, DMm);
        // causal conv + SiLU
        dm_conv_silu<<<(ROWS * DIm) / 256, 256, 0, stream>>>(
            xz, conv_w + (size_t)i * DIm * DCc, conv_b + (size_t)i * DIm,
            xc32, xc16);
        // xproj: (2048,1024) x (64,1024)^T -> (2048,64)
        dm_gemm_wmma<0><<<dim3(1, ROWS / 16), 32, 0, stream>>>(
            xc16, DIm, w16xp + (size_t)i * (DTRr + 2 * DSs) * DIm, DIm,
            xdbl, DTRr + 2 * DSs, nullptr, DIm);
        // dt_low extract -> f16
        dm_dtlow<<<(ROWS * DTRr) / 256, 256, 0, stream>>>(xdbl, dtlo16);
        // dtproj + bias + softplus: (2048,32) x (1024,32)^T -> (2048,1024)
        dm_gemm_wmma<1><<<dim3(DIm / 64, ROWS / 16), 32, 0, stream>>>(
            dtlo16, DTRr, w16dt + (size_t)i * DIm * DTRr, DTRr,
            dtb, DIm, dtproj_b + (size_t)i * DIm, DTRr);
        // selective scan + D*u + SiLU(z) gating
        dm_scan<<<BB * (DIm / 256), 256, 0, stream>>>(
            dtb, xdbl, xc32, xz,
            A_log + (size_t)i * DIm * DSs, D_ssm + (size_t)i * DIm, yg16);
        // out_proj + residual: x += (2048,1024) x (512,1024)^T
        dm_gemm_wmma<2><<<dim3(DMm / 64, ROWS / 16), 32, 0, stream>>>(
            yg16, DIm, w16out + (size_t)i * DMm * DIm, DIm,
            x, DMm, nullptr, DIm);
    }

    // final LayerNorm + heads
    dm_layernorm<<<ROWS, 256, 0, stream>>>(x, fn_g, fn_b, xf32, xn16);
    dm_heads<<<ROWS, 32, 0, stream>>>(xf32, W_ps, b_ps, W_pa, b_pa, (float*)d_out);
}
